// Linear_MAE_6562710028357
// MI455X (gfx1250) — matrix operand order repaired and checked by reading, not compiled, not run on hardware
//
#include <hip/hip_runtime.h>
#include <math.h>

// ---------------------------------------------------------------------------
// Problem constants (from the reference): B=16, T=1024 -> BT=16384 rows
//   spikes: (BT,1024) f32, R=16 areas x NP=64, NC=32, H=256, NL=20 -> 320 lat
// ---------------------------------------------------------------------------
#define BT   16384
#define NIN  1024     // R*NP
#define HID  256
#define RNL  320      // R*NL

typedef _Float16 half_t;
typedef __attribute__((ext_vector_type(16))) _Float16 v16h;
typedef __attribute__((ext_vector_type(8)))  float    v8f;

#define WMMA_F32_F16(a, b, c) \
  __builtin_amdgcn_wmma_f32_16x16x32_f16(false, (a), false, (b), (short)0, (c), false, false)

// ---- CDNA5 async global->LDS copies (ASYNCcnt-tracked DMA, ISA 15.18.3) ----
__device__ __forceinline__ unsigned lds_off32(const void* p) {
  // Flat LDS aperture: addr[31:0] is the wave-relative LDS byte offset.
  return (unsigned)(unsigned long long)p;
}
__device__ __forceinline__ void async_b128(unsigned lds_dst, const void* src) {
  asm volatile("global_load_async_to_lds_b128 %0, %1, off"
               :: "v"(lds_dst), "v"(src) : "memory");
}
__device__ __forceinline__ void async_b64(unsigned lds_dst, const void* src) {
  asm volatile("global_load_async_to_lds_b64 %0, %1, off"
               :: "v"(lds_dst), "v"(src) : "memory");
}

// ---------------------------------------------------------------------------
// Weight-prep kernels: fold stitcher+mask into W_U -> W_eff^T (K-major rows so
// async DMA can copy B tiles verbatim), b_eff; W_V^T in f16; W_dec K-pad 20->32.
// ---------------------------------------------------------------------------
__global__ void prep_weff(const float* __restrict__ Wst, const float* __restrict__ Wu,
                          const int* __restrict__ keep, half_t* __restrict__ WeffT) {
  int idx = blockIdx.x * blockDim.x + threadIdx.x;       // 0 .. 1024*256-1
  if (idx >= NIN * HID) return;
  int k = idx >> 8;                                      // input neuron 0..1023
  int n = idx & 255;                                     // hidden unit
  int r = k >> 6;                                        // area
  float s = 0.f;
  if (keep[r]) {
    #pragma unroll 4
    for (int c = 0; c < 32; ++c)
      s += Wst[(size_t)k * 32 + c] * Wu[(size_t)(r * 32 + c) * HID + n];
  }
  WeffT[(size_t)n * NIN + k] = (half_t)s;                // transposed [n][k]
}

__global__ void prep_beff(const float* __restrict__ bst, const float* __restrict__ Wu,
                          const float* __restrict__ bu, const int* __restrict__ keep,
                          float* __restrict__ beff) {
  int n = threadIdx.x;                                   // 0..255
  float s = bu[n];
  for (int r = 0; r < 16; ++r) {
    if (!keep[r]) continue;
    for (int c = 0; c < 32; ++c)
      s += bst[r * 32 + c] * Wu[(size_t)(r * 32 + c) * HID + n];
  }
  beff[n] = s;
}

__global__ void prep_wv(const float* __restrict__ Wv, half_t* __restrict__ WvT) {
  int i = blockIdx.x * blockDim.x + threadIdx.x;         // over (k,n), 256*320
  if (i >= HID * RNL) return;
  int k = i / RNL, n = i - k * RNL;
  WvT[(size_t)n * HID + k] = (half_t)Wv[i];              // transposed [n][k]
}

__global__ void prep_wd(const float* __restrict__ Wd, half_t* __restrict__ out) {
  int i = blockIdx.x * blockDim.x + threadIdx.x;         // 16*32*64
  if (i >= 16 * 32 * 64) return;
  int n = i & 63;
  int k = (i >> 6) & 31;
  int r = i >> 11;
  out[i] = (half_t)((k < 20) ? Wd[((size_t)r * 20 + k) * 64 + n] : 0.f);
}

__global__ void zero_acc(float* acc) { if (threadIdx.x < 8) acc[threadIdx.x] = 0.f; }

// ---------------------------------------------------------------------------
// GEMM 1: h(BT,256) = f16( spikes(BT,1024) @ W_eff(1024,256) + b_eff )
// 8 waves, block tile 64(M) x 32(N), K stepped by 32.
// Double-buffered async global->LDS staging: per thread per stage
//   2x async b128 (A tile, raw f32) + 1x async b64 (B tile, f16) = 3 in flight.
// ---------------------------------------------------------------------------
__global__ __launch_bounds__(256) void gemm_stitchU(
    const float* __restrict__ X, const half_t* __restrict__ WT,
    const float* __restrict__ bias, half_t* __restrict__ Hout) {
  __shared__ float  Af[2][64 * 32];   // [m][k] raw f32, 8 KB per stage
  __shared__ half_t Bh[2][32 * 32];   // [n][k] f16,     2 KB per stage
  const int tid  = threadIdx.x;
  const int lane = tid & 31, wave = tid >> 5;
  const int wm = wave & 3, wn = wave >> 2;          // 4x2 wave grid
  const int m0 = blockIdx.x * 64, n0 = blockIdx.y * 32;
  const int hi = lane >> 4, lm = lane & 15;

  const int arow = tid >> 2, aseg = tid & 3;        // A: 4 thr/row, 2x16B each
  const int brow = tid >> 3, bseg = tid & 7;        // B: 8 thr/row, 1x8B each
  const float*  asrc = X  + (size_t)(m0 + arow) * NIN + aseg * 8;
  const half_t* bsrc = WT + (size_t)(n0 + brow) * NIN + bseg * 4;

  #define G1_ISSUE(stage, k0)                                             \
    do {                                                                  \
      async_b128(lds_off32(&Af[stage][arow * 32 + aseg * 8 + 0]), asrc + (k0) + 0); \
      async_b128(lds_off32(&Af[stage][arow * 32 + aseg * 8 + 4]), asrc + (k0) + 4); \
      async_b64 (lds_off32(&Bh[stage][brow * 32 + bseg * 4]),     bsrc + (k0));     \
    } while (0)

  v8f acc = {};
  const int nIter = NIN / 32;
  G1_ISSUE(0, 0);
  for (int i = 0; i < nIter - 1; ++i) {
    G1_ISSUE((i + 1) & 1, (i + 1) * 32);
    __builtin_amdgcn_s_wait_asynccnt(3);   // stage i landed (in-order)
    __syncthreads();
    {
      const float*  ar = &Af[i & 1][(wm * 16 + lm) * 32];
      const half_t* br = &Bh[i & 1][(wn * 16 + lm) * 32];
      v16h a, b;
      #pragma unroll
      for (int j = 0; j < 16; ++j) {
        int ka = hi * 16 + j;                      // contiguous k layout (matches B)
        a[j] = (half_t)ar[ka];
        b[j] = br[hi * 16 + j];                    // contiguous k layout (matches A)
      }
      acc = WMMA_F32_F16(a, b, acc);
    }
    __syncthreads();                       // readers done before buffer reuse
  }
  __builtin_amdgcn_s_wait_asynccnt(0);
  __syncthreads();
  {
    const int s = (nIter - 1) & 1;
    const float*  ar = &Af[s][(wm * 16 + lm) * 32];
    const half_t* br = &Bh[s][(wn * 16 + lm) * 32];
    v16h a, b;
    #pragma unroll
    for (int j = 0; j < 16; ++j) {
      int ka = hi * 16 + j;
      a[j] = (half_t)ar[ka];
      b[j] = br[hi * 16 + j];
    }
    acc = WMMA_F32_F16(a, b, acc);
  }
  #pragma unroll
  for (int v = 0; v < 8; ++v) {
    int m = m0 + wm * 16 + v + hi * 8;
    int n = n0 + wn * 16 + lm;
    Hout[(size_t)m * HID + n] = (half_t)(acc[v] + bias[n]);
  }
  #undef G1_ISSUE
}

// ---------------------------------------------------------------------------
// GEMM 2: lat(BT,320) = h(BT,256) @ W_V(256,320) + b_V   (f32 out for reg loss)
// All-f16 staging: 1x async b128 (A) + 1x async b64 (B) per thread per stage.
// ---------------------------------------------------------------------------
__global__ __launch_bounds__(256) void gemm_V(
    const half_t* __restrict__ Hin, const half_t* __restrict__ WT,
    const float* __restrict__ bias, float* __restrict__ Lat) {
  __shared__ half_t Ah[2][64 * 32];   // [m][k]
  __shared__ half_t Bh[2][32 * 32];   // [n][k]
  const int tid  = threadIdx.x;
  const int lane = tid & 31, wave = tid >> 5;
  const int wm = wave & 3, wn = wave >> 2;
  const int m0 = blockIdx.x * 64, n0 = blockIdx.y * 32;
  const int hi = lane >> 4, lm = lane & 15;

  const int arow = tid >> 2, aseg = tid & 3;        // A: 4 thr/row, 1x16B each
  const int brow = tid >> 3, bseg = tid & 7;        // B: 8 thr/row, 1x8B each
  const half_t* asrc = Hin + (size_t)(m0 + arow) * HID + aseg * 8;
  const half_t* bsrc = WT  + (size_t)(n0 + brow) * HID + bseg * 4;

  #define G2_ISSUE(stage, k0)                                             \
    do {                                                                  \
      async_b128(lds_off32(&Ah[stage][arow * 32 + aseg * 8]), asrc + (k0)); \
      async_b64 (lds_off32(&Bh[stage][brow * 32 + bseg * 4]), bsrc + (k0)); \
    } while (0)

  v8f acc = {};
  const int nIter = HID / 32;
  G2_ISSUE(0, 0);
  for (int i = 0; i < nIter - 1; ++i) {
    G2_ISSUE((i + 1) & 1, (i + 1) * 32);
    __builtin_amdgcn_s_wait_asynccnt(2);
    __syncthreads();
    {
      const half_t* ar = &Ah[i & 1][(wm * 16 + lm) * 32];
      const half_t* br = &Bh[i & 1][(wn * 16 + lm) * 32];
      v16h a, b;
      #pragma unroll
      for (int j = 0; j < 16; ++j) {
        int ka = hi * 16 + j;
        a[j] = ar[ka];
        b[j] = br[hi * 16 + j];
      }
      acc = WMMA_F32_F16(a, b, acc);
    }
    __syncthreads();
  }
  __builtin_amdgcn_s_wait_asynccnt(0);
  __syncthreads();
  {
    const int s = (nIter - 1) & 1;
    const half_t* ar = &Ah[s][(wm * 16 + lm) * 32];
    const half_t* br = &Bh[s][(wn * 16 + lm) * 32];
    v16h a, b;
    #pragma unroll
    for (int j = 0; j < 16; ++j) {
      int ka = hi * 16 + j;
      a[j] = ar[ka];
      b[j] = br[hi * 16 + j];
    }
    acc = WMMA_F32_F16(a, b, acc);
  }
  #pragma unroll
  for (int v = 0; v < 8; ++v) {
    int m = m0 + wm * 16 + v + hi * 8;
    int n = n0 + wn * 16 + lm;
    Lat[(size_t)m * RNL + n] = acc[v] + bias[n];
  }
  #undef G2_ISSUE
}

// ---------------------------------------------------------------------------
// GEMM 3 (fused): per (16-row tile, area r): preds = lat_r @ W_dec[r] + b_dec[r]
// K padded 20->32; accumulates Poisson NLL sum(exp(p) - t*p).
// ---------------------------------------------------------------------------
__global__ __launch_bounds__(32) void dec_poisson(
    const float* __restrict__ Lat, const half_t* __restrict__ Wd,
    const float* __restrict__ bdec, const float* __restrict__ X,
    float* __restrict__ preds, float* __restrict__ acc) {
  const int lane = threadIdx.x;
  const int rt = blockIdx.x;           // row tile 0..1023
  const int r  = blockIdx.y;           // area 0..15
  const int hi = lane >> 4, lm = lane & 15;

  v16h a;
  {
    const float* lrow = Lat + (size_t)(rt * 16 + lm) * RNL + r * 20;
    #pragma unroll
    for (int j = 0; j < 16; ++j) {
      int k = hi * 16 + j;
      a[j] = (half_t)((k < 20) ? lrow[k] : 0.f);
    }
  }
  const half_t* Wr = Wd + r * (32 * 64);
  float psum = 0.f;
  #pragma unroll
  for (int nt = 0; nt < 4; ++nt) {
    v16h b;
    #pragma unroll
    for (int j = 0; j < 16; ++j) b[j] = Wr[(hi * 16 + j) * 64 + nt * 16 + lm];
    v8f c = {};
    c = WMMA_F32_F16(a, b, c);
    #pragma unroll
    for (int v = 0; v < 8; ++v) {
      int m  = rt * 16 + v + hi * 8;
      int nl = nt * 16 + lm;
      int ng = r * 64 + nl;
      float p = c[v] + bdec[r * 64 + nl];
      preds[(size_t)m * NIN + ng] = p;
      psum += __expf(p) - X[(size_t)m * NIN + ng] * p;
    }
  }
  #pragma unroll
  for (int off = 16; off > 0; off >>= 1) psum += __shfl_xor(psum, off, 32);
  if (lane == 0) atomicAdd(acc, psum);
}

// ---------------------------------------------------------------------------
// reg_loss: sum |lat[b,t+1,f] - lat[b,t,f]| over 16 x 1023 x 320
// ---------------------------------------------------------------------------
__global__ __launch_bounds__(256) void reg_loss_kernel(const float* __restrict__ Lat,
                                                       float* __restrict__ acc) {
  __shared__ float sdata[256];
  const long NTOT = 16L * 1023L * 320L;
  float s = 0.f;
  for (long i = (long)blockIdx.x * 256 + threadIdx.x; i < NTOT;
       i += (long)gridDim.x * 256) {
    long b   = i / (1023L * 320L);
    long rem = i - b * (1023L * 320L);
    long t   = rem / 320L;
    long f   = rem - t * 320L;
    size_t base = ((size_t)b * 1024 + (size_t)t) * RNL + (size_t)f;
    s += fabsf(Lat[base + RNL] - Lat[base]);
  }
  sdata[threadIdx.x] = s;
  __syncthreads();
  for (int off = 128; off > 0; off >>= 1) {
    if (threadIdx.x < (unsigned)off) sdata[threadIdx.x] += sdata[threadIdx.x + off];
    __syncthreads();
  }
  if (threadIdx.x == 0) atomicAdd(acc + 1, sdata[0]);
}

__global__ void finalize_kernel(const float* __restrict__ acc, float* __restrict__ out) {
  out[0] = acc[0] / ((float)BT * (float)NIN);
  out[1] = acc[1] / (16.f * 1023.f * 320.f) * 0.1f;
}

// ---------------------------------------------------------------------------
extern "C" void kernel_launch(void* const* d_in, const int* in_sizes, int n_in,
                              void* d_out, int out_size, void* d_ws, size_t ws_size,
                              hipStream_t stream) {
  (void)in_sizes; (void)n_in; (void)out_size; (void)ws_size;
  const float* spikes = (const float*)d_in[0];
  // d_in[1] = neuron_regions (contiguous layout is baked in)
  const int*   keep   = (const int*)d_in[2];
  const float* Wst    = (const float*)d_in[3];
  const float* bst    = (const float*)d_in[4];
  const float* Wu     = (const float*)d_in[5];
  const float* bu     = (const float*)d_in[6];
  const float* Wv     = (const float*)d_in[7];
  const float* bv     = (const float*)d_in[8];
  const float* Wd     = (const float*)d_in[9];
  const float* bd     = (const float*)d_in[10];

  char* ws = (char*)d_ws;
  size_t o = 0;
  auto take = [&](size_t bytes) -> char* {
    char* p = ws + o;
    o = (o + bytes + 255) & ~(size_t)255;
    return p;
  };
  float*  acc   = (float*)take(32);
  half_t* WeffT = (half_t*)take((size_t)NIN * HID * sizeof(half_t));   // 512 KB
  float*  beff  = (float*)take(HID * sizeof(float));
  half_t* WvT   = (half_t*)take((size_t)HID * RNL * sizeof(half_t));   // 160 KB
  half_t* Wd16  = (half_t*)take((size_t)16 * 32 * 64 * sizeof(half_t));// 64 KB
  half_t* Hbuf  = (half_t*)take((size_t)BT * HID * sizeof(half_t));    // 8 MB
  float*  Lat   = (float*)take((size_t)BT * RNL * sizeof(float));      // 21 MB

  float* out   = (float*)d_out;
  float* preds = out + 2;

  zero_acc<<<1, 32, 0, stream>>>(acc);
  prep_weff<<<(NIN * HID + 255) / 256, 256, 0, stream>>>(Wst, Wu, keep, WeffT);
  prep_beff<<<1, 256, 0, stream>>>(bst, Wu, bu, keep, beff);
  prep_wv<<<(HID * RNL + 255) / 256, 256, 0, stream>>>(Wv, WvT);
  prep_wd<<<(16 * 32 * 64 + 255) / 256, 256, 0, stream>>>(Wd, Wd16);

  gemm_stitchU<<<dim3(BT / 64, HID / 32), 256, 0, stream>>>(spikes, WeffT, beff, Hbuf);
  gemm_V<<<dim3(BT / 64, RNL / 32), 256, 0, stream>>>(Hbuf, WvT, bv, Lat);
  dec_poisson<<<dim3(BT / 16, 16), 32, 0, stream>>>(Lat, Wd16, bd, spikes, preds, acc);
  reg_loss_kernel<<<2048, 256, 0, stream>>>(Lat, acc);
  finalize_kernel<<<1, 1, 0, stream>>>(acc, out);
}
